// serverGAT_21964462752545
// MI455X (gfx1250) — compile-verified, matrix operand order checked by reading, not run on hardware
//
#include <hip/hip_runtime.h>
#include <math.h>

// ---------------- CDNA5 vector types ----------------
typedef __attribute__((ext_vector_type(16))) __bf16        bf16x16;
typedef __attribute__((ext_vector_type(8)))  float         f32x8;
typedef __attribute__((ext_vector_type(4)))  unsigned int  u32x4;
typedef __attribute__((ext_vector_type(4)))  float         f32x4;

// fp32 -> bf16 round-to-nearest-even
__device__ __forceinline__ unsigned short f2bf(float f) {
    unsigned int u = __float_as_uint(f);
    unsigned int r = 0x7FFFu + ((u >> 16) & 1u);
    return (unsigned short)((u + r) >> 16);
}

// ---------------- convert f32 matrix -> bf16 (K-contiguous) ----------------
__global__ void conv_bf16_kernel(const float* __restrict__ x,
                                 unsigned short* __restrict__ y, size_t n) {
    size_t i = (size_t)blockIdx.x * blockDim.x + threadIdx.x;
    if (i < n) y[i] = f2bf(x[i]);
}

// ---------------- transpose weight [K,Nout] -> bf16 [Npad,K], zero-padded ---
__global__ void transpose_w_kernel(const float* __restrict__ W,
                                   unsigned short* __restrict__ Wt,
                                   int K, int Nout, int Npad) {
    int i = blockIdx.x * blockDim.x + threadIdx.x;
    if (i >= K * Npad) return;
    int n = i / K, k = i - n * K;
    Wt[i] = (n < Nout) ? f2bf(W[(size_t)k * Nout + n]) : (unsigned short)0;
}

// ---------------- WMMA GEMM: C[M,nvalid] = Abf[M,K] @ Bt[Npad,K]^T ----------
// One wave computes a 16x(16*NB) strip: one A fragment reused by NB WMMAs.
// All K-step loads (A pair + NB B pairs) are issued before the WMMA burst so
// the backend can use partial s_wait_loadcnt and keep WMMAs back-to-back.
// Abf row-major [M,K] bf16; Bt row-major [Npad,K] bf16 (W transposed).
// Lane layout per ISA: lanes 0-15 carry K {0..7,16..23}, lanes 16-31 carry
// K {8..15,24..31}; C/D: VGPR r -> row (r or r+8), N = lane&15.
template <int NB>
__global__ void gemm_wmma_bf16(const unsigned short* __restrict__ Abf,
                               const unsigned short* __restrict__ Bt,
                               const float* __restrict__ bias,
                               float* __restrict__ Cf32,
                               unsigned short* __restrict__ Cbf,
                               int Mtiles, int NtGroups, int K, int ldc,
                               int nvalid, int act) {
    const int w = (int)((blockIdx.x * blockDim.x + threadIdx.x) >> 5);
    if (w >= Mtiles * NtGroups) return;          // whole-wave uniform exit
    const int lane = threadIdx.x & 31;
    const int tm  = w / NtGroups;
    const int tn0 = (w - tm * NtGroups) * NB;    // first N tile of this strip
    const int l15 = lane & 15;
    const int kh  = (lane >> 4) << 3;            // 0 or 8
    const size_t arow  = (size_t)(tm * 16 + l15) * (size_t)K;
    const size_t brow0 = (size_t)(tn0 * 16 + l15) * (size_t)K;

    f32x8 acc[NB];
#pragma unroll
    for (int j = 0; j < NB; ++j) acc[j] = (f32x8){0,0,0,0,0,0,0,0};

    union Frag { u32x4 u[2]; bf16x16 v; };

    for (int kt = 0; kt < K; kt += 32) {
        Frag a, b[NB];
        // issue ALL loads for this K step first (A + NB B fragments)
        a.u[0] = *(const u32x4*)(Abf + arow + kt + kh);
        a.u[1] = *(const u32x4*)(Abf + arow + kt + kh + 16);
#pragma unroll
        for (int j = 0; j < NB; ++j) {
            const size_t brow = brow0 + (size_t)j * 16 * K;
            b[j].u[0] = *(const u32x4*)(Bt + brow + kt + kh);
            b[j].u[1] = *(const u32x4*)(Bt + brow + kt + kh + 16);
        }
        // WMMA burst: one A operand reused NB times
#pragma unroll
        for (int j = 0; j < NB; ++j) {
            acc[j] = __builtin_amdgcn_wmma_f32_16x16x32_bf16(
                         false, a.v, false, b[j].v, (short)0, acc[j],
                         false, false);
        }
    }

    const int rbase = tm * 16 + ((lane >> 4) << 3);
#pragma unroll
    for (int j = 0; j < NB; ++j) {
        const int col = (tn0 + j) * 16 + l15;
        if (col >= nvalid) continue;
        const float bv = bias ? bias[col] : 0.0f;
#pragma unroll
        for (int r = 0; r < 8; ++r) {
            float v = acc[j][r] + bv;
            if (act) v = v > 0.f ? v : 0.f;
            const size_t idx = (size_t)(rbase + r) * (size_t)ldc + col;
            if (Cf32) Cf32[idx] = v;
            if (Cbf)  Cbf[idx]  = f2bf(v);
        }
    }
}

// ---------------- per-node dual dot products: s = h . a ---------------------
__global__ void row_dots_kernel(const float* __restrict__ Hm,
                                const float* __restrict__ a_s,
                                const float* __restrict__ a_d,
                                float* __restrict__ s_src,
                                float* __restrict__ s_dst, int Nn) {
    int wv = (int)((blockIdx.x * blockDim.x + threadIdx.x) >> 5);
    if (wv >= Nn) return;
    int lane = threadIdx.x & 31;
    const float* hp = Hm + (size_t)wv * 256 + lane * 8;
    const float* ap = a_s + lane * 8;
    const float* dp = a_d + lane * 8;
    float ss = 0.f, sd = 0.f;
#pragma unroll
    for (int j = 0; j < 8; ++j) { float h = hp[j]; ss += h * ap[j]; sd += h * dp[j]; }
#pragma unroll
    for (int off = 16; off > 0; off >>= 1) {
        ss += __shfl_xor(ss, off, 32);
        sd += __shfl_xor(sd, off, 32);
    }
    if (lane == 0) { s_src[wv] = ss; s_dst[wv] = sd; }
}

// ---------------- init accumulator / segment-max / segment-sum --------------
__global__ void init_agg_kernel(float* __restrict__ Out, float* __restrict__ m,
                                float* __restrict__ denom, int Nn) {
    size_t i = (size_t)blockIdx.x * blockDim.x + threadIdx.x;
    if (i < (size_t)Nn * 256) Out[i] = 0.f;
    if (i < (size_t)Nn) { m[i] = __uint_as_float(0xFF800000u); denom[i] = 0.f; }
}

__device__ __forceinline__ void edge_nodes(const int* __restrict__ srcA,
                                           const int* __restrict__ dstA,
                                           long long e, long long E, int Nn,
                                           int& s, int& d) {
    if (e < E) {
        s = srcA[e]; d = dstA[e];
        if ((unsigned)s >= (unsigned)Nn) s = 0;   // defensive clamp
        if ((unsigned)d >= (unsigned)Nn) d = 0;
    } else {
        s = d = (int)(e - E);                     // self-loop
    }
}

// score = leaky_relu(s_src[src] + s_dst[dst]); segment-max into m[dst]
__global__ void edge_score_max_kernel(const int* __restrict__ srcA,
                                      const int* __restrict__ dstA,
                                      const float* __restrict__ s_src,
                                      const float* __restrict__ s_dst,
                                      float* __restrict__ score,
                                      float* __restrict__ m,
                                      long long E, long long Etot, int Nn) {
    long long e = (long long)blockIdx.x * blockDim.x + threadIdx.x;
    if (e >= Etot) return;
    int s, d; edge_nodes(srcA, dstA, e, E, Nn, s, d);
    float sc = s_src[s] + s_dst[d];
    sc = sc > 0.f ? sc : 0.2f * sc;
    score[e] = sc;
    // float atomic max via sign-split trick (init is -inf)
    if (sc >= 0.f) atomicMax((int*)(m + d), __float_as_int(sc));
    else           atomicMin((unsigned int*)(m + d), __float_as_uint(sc));
}

// e = exp(score - m[dst]); segment-sum into denom[dst]
__global__ void edge_exp_sum_kernel(const int* __restrict__ srcA,
                                    const int* __restrict__ dstA,
                                    const float* __restrict__ m,
                                    float* __restrict__ score,
                                    float* __restrict__ denom,
                                    long long E, long long Etot, int Nn) {
    long long e = (long long)blockIdx.x * blockDim.x + threadIdx.x;
    if (e >= Etot) return;
    int s, d; edge_nodes(srcA, dstA, e, E, Nn, s, d);
    float ex = __expf(score[e] - m[d]);
    score[e] = ex;
    atomicAdd(denom + d, ex);
}

// Out[dst,:] += h[src,:] * (e / denom[dst]); one wave per edge, 8 f32/lane
__global__ void edge_aggregate_kernel(const int* __restrict__ srcA,
                                      const int* __restrict__ dstA,
                                      const float* __restrict__ score,
                                      const float* __restrict__ denom,
                                      const float* __restrict__ Hm,
                                      float* __restrict__ Out,
                                      long long E, long long Etot, int Nn) {
    long long e = ((long long)blockIdx.x * blockDim.x + threadIdx.x) >> 5;
    if (e >= Etot) return;
    int lane = threadIdx.x & 31;
    int s, d; edge_nodes(srcA, dstA, e, E, Nn, s, d);
    float alpha = score[e] / (denom[d] + 1e-16f);
    const f32x4* hp = (const f32x4*)(Hm + (size_t)s * 256 + lane * 8);
    f32x4 h0 = hp[0], h1 = hp[1];
    float* op = Out + (size_t)d * 256 + lane * 8;
    atomicAdd(op + 0, h0.x * alpha);
    atomicAdd(op + 1, h0.y * alpha);
    atomicAdd(op + 2, h0.z * alpha);
    atomicAdd(op + 3, h0.w * alpha);
    atomicAdd(op + 4, h1.x * alpha);
    atomicAdd(op + 5, h1.y * alpha);
    atomicAdd(op + 6, h1.z * alpha);
    atomicAdd(op + 7, h1.w * alpha);
}

// -------- bias + optional relu (in place) + optional fused bf16 emit --------
__global__ void bias_act_kernel(float* __restrict__ v, const float* __restrict__ b,
                                unsigned short* __restrict__ vbf,
                                int Nn, int act) {
    size_t i = (size_t)blockIdx.x * blockDim.x + threadIdx.x;
    if (i >= (size_t)Nn * 256) return;
    float x = v[i] + b[i & 255];
    if (act) x = x > 0.f ? x : 0.f;
    v[i] = x;
    if (vbf) vbf[i] = f2bf(x);
}

// =============================== launcher ===================================
extern "C" void kernel_launch(void* const* d_in, const int* in_sizes, int n_in,
                              void* d_out, int out_size, void* d_ws, size_t ws_size,
                              hipStream_t stream) {
    const int  F = 256, Hh = 256, C = 40, CPAD = 48;
    const int  N    = in_sizes[0] / F;         // 50000 (divisible by 16)
    const long long E    = in_sizes[1] / 2;    // 800000
    const long long Etot = E + N;              // + self loops

    const float* X   = (const float*)d_in[0];
    const int*   src = (const int*)d_in[1];
    const int*   dst = src + E;

    const float* W[3]  = {(const float*)d_in[2], (const float*)d_in[6],  (const float*)d_in[10]};
    const float* AS[3] = {(const float*)d_in[3], (const float*)d_in[7],  (const float*)d_in[11]};
    const float* AD[3] = {(const float*)d_in[4], (const float*)d_in[8],  (const float*)d_in[12]};
    const float* B[3]  = {(const float*)d_in[5], (const float*)d_in[9],  (const float*)d_in[13]};
    const float* Wp = (const float*)d_in[14];
    const float* bp = (const float*)d_in[15];
    const float* Wr = (const float*)d_in[16];
    const float* br = (const float*)d_in[17];

    // ---- workspace carve-out (all read buffers initialized every call) ----
    char* p = (char*)d_ws;
    auto carve = [&](size_t bytes) -> void* {
        void* r = (void*)p; p += (bytes + 255) & ~(size_t)255; return r;
    };
    float*          Hlin = (float*)         carve((size_t)N * Hh * 4);
    float*          Agg  = (float*)         carve((size_t)N * Hh * 4);
    unsigned short* Abf  = (unsigned short*)carve((size_t)N * Hh * 2);
    unsigned short* Wt   = (unsigned short*)carve((size_t)Hh * Hh * 2);
    float*          sS   = (float*)         carve((size_t)N * 4);
    float*          sD   = (float*)         carve((size_t)N * 4);
    float*          mB   = (float*)         carve((size_t)N * 4);
    float*          dB   = (float*)         carve((size_t)N * 4);
    float*          sc   = (float*)         carve((size_t)Etot * 4);

    const unsigned TB = 256;
    const unsigned gNH = (unsigned)(((size_t)N * Hh + TB - 1) / TB);
    const unsigned gE  = (unsigned)((Etot + TB - 1) / TB);
    const unsigned gEw = (unsigned)((Etot + 7) / 8);       // 1 wave / edge
    const unsigned gNd = (unsigned)((N + 7) / 8);          // 1 wave / node
    const int Mtiles = N / 16;

    // 256-wide GEMM: NB=4 -> 4 N-tiles per wave, 4 strip-groups
    auto gemm256 = [&](const unsigned short* a, const float* bias, float* cf,
                       unsigned short* cbf, int act) {
        const int ng = (Hh / 16) / 4;                       // 4
        unsigned blocks = (unsigned)((Mtiles * ng + 7) / 8);
        gemm_wmma_bf16<4><<<blocks, TB, 0, stream>>>(a, Wt, bias, cf, cbf,
                                                     Mtiles, ng, F, Hh, Hh, act);
    };

    auto gat_layer = [&](const float* Wl, const float* as, const float* ad,
                         const float* bl, int relu) {
        transpose_w_kernel<<<(Hh * Hh + TB - 1) / TB, TB, 0, stream>>>(Wl, Wt, F, Hh, Hh);
        gemm256(Abf, nullptr, Hlin, nullptr, 0);            // h = x @ W
        row_dots_kernel<<<gNd, TB, 0, stream>>>(Hlin, as, ad, sS, sD, N);
        init_agg_kernel<<<gNH, TB, 0, stream>>>(Agg, mB, dB, N);
        edge_score_max_kernel<<<gE, TB, 0, stream>>>(src, dst, sS, sD, sc, mB, E, Etot, N);
        edge_exp_sum_kernel<<<gE, TB, 0, stream>>>(src, dst, mB, sc, dB, E, Etot, N);
        edge_aggregate_kernel<<<gEw, TB, 0, stream>>>(src, dst, sc, dB, Hlin, Agg, E, Etot, N);
        // bias (+relu) and fused bf16 emit for the NEXT GEMM's A operand
        bias_act_kernel<<<gNH, TB, 0, stream>>>(Agg, bl, Abf, N, relu);
    };

    // initial x -> bf16 (only explicit conversion pass)
    conv_bf16_kernel<<<gNH, TB, 0, stream>>>(X, Abf, (size_t)N * Hh);

    // three GAT layers (relu, relu, none); each leaves Abf = bf16(layer out)
    gat_layer(W[0], AS[0], AD[0], B[0], 1);
    gat_layer(W[1], AS[1], AD[1], B[1], 1);
    gat_layer(W[2], AS[2], AD[2], B[2], 0);

    // post1: relu(h @ Wp + bp); epilogue emits bf16 directly into Abf
    transpose_w_kernel<<<(Hh * Hh + TB - 1) / TB, TB, 0, stream>>>(Wp, Wt, F, Hh, Hh);
    gemm256(Abf, bp, Hlin, Abf, 1);

    // readout: h @ Wr + br -> d_out [N, 40] (3 padded tiles in one wave)
    transpose_w_kernel<<<(CPAD * F + TB - 1) / TB, TB, 0, stream>>>(Wr, Wt, F, C, CPAD);
    {
        unsigned blocks = (unsigned)((Mtiles * 1 + 7) / 8);
        gemm_wmma_bf16<3><<<blocks, TB, 0, stream>>>(Abf, Wt, br, (float*)d_out,
                                                     nullptr, Mtiles, 1, F, C, C, 0);
    }
}
